// GAT_38130719654220
// MI455X (gfx1250) — compile-verified, matrix operand order checked by reading, not run on hardware
//
#include <hip/hip_runtime.h>
#include <math.h>

// ---------------------------------------------------------------------------
// Types for CDNA5 WMMA (wave32): v_wmma_f32_16x16x32_bf16
// ---------------------------------------------------------------------------
typedef __attribute__((ext_vector_type(16))) __bf16 v16bf;
typedef __attribute__((ext_vector_type(8)))  __bf16 v8bf;
typedef __attribute__((ext_vector_type(8)))  float  v8f;

// ---------------------------------------------------------------------------
// GEMM: C[M,Nout] = A[M,K](fp32 -> bf16) * W[K,Nout](fp32 -> bf16, staged in
// LDS transposed), fp32 accumulate via V_WMMA_F32_16X16X32_BF16.
// blockDim = 256 (8 waves); each wave owns one 16-row M tile and all N tiles.
// A-fragment layout (ISA 7.12.2, 16-bit A 16x32): lanes 0-15 hold row M=lane,
// K = {0..7, 16..23}; lanes 16-31 hold row M=lane-16, K = {8..15, 24..31}.
// B symmetric with N in place of M. C/D: vgpr v -> row v + (lane<16?0:8),
// col = lane&15.
// ---------------------------------------------------------------------------
template <int KT, int NT>
__global__ __launch_bounds__(256) void wmma_gemm(
    const float* __restrict__ A, const float* __restrict__ W,
    const float* __restrict__ bias, float* __restrict__ C,
    int M, int relu)
{
    constexpr int K = KT * 32;
    constexpr int Nout = NT * 16;
    __shared__ alignas(16) __bf16 sW[Nout * K];   // W transposed: sW[n*K + k]

    const int tid = threadIdx.x;
    for (int i = tid; i < K * Nout; i += 256) {
        int k = i / Nout;
        int n = i - k * Nout;
        sW[n * K + k] = (__bf16)W[i];
    }
    __syncthreads();

    const int wave = tid >> 5;
    const int lane = tid & 31;
    const int row0 = blockIdx.x * 128 + wave * 16;
    if (row0 >= M) return;

    const int hl  = lane & 15;
    const int off = (lane < 16) ? 0 : 8;
    const float* pA = A + (size_t)(row0 + hl) * K;

    v8f acc[NT] = {};

#pragma unroll
    for (int kb = 0; kb < KT; ++kb) {
        const int kb0 = kb * 32;
        if (kb + 1 < KT)
            __builtin_prefetch(pA + kb0 + 32, 0, 0);   // global_prefetch_b8

        float4 a0 = *(const float4*)(pA + kb0 + off);
        float4 a1 = *(const float4*)(pA + kb0 + off + 4);
        float4 a2 = *(const float4*)(pA + kb0 + 16 + off);
        float4 a3 = *(const float4*)(pA + kb0 + 16 + off + 4);
        v16bf a;
        a[0]  = (__bf16)a0.x; a[1]  = (__bf16)a0.y; a[2]  = (__bf16)a0.z; a[3]  = (__bf16)a0.w;
        a[4]  = (__bf16)a1.x; a[5]  = (__bf16)a1.y; a[6]  = (__bf16)a1.z; a[7]  = (__bf16)a1.w;
        a[8]  = (__bf16)a2.x; a[9]  = (__bf16)a2.y; a[10] = (__bf16)a2.z; a[11] = (__bf16)a2.w;
        a[12] = (__bf16)a3.x; a[13] = (__bf16)a3.y; a[14] = (__bf16)a3.z; a[15] = (__bf16)a3.w;

#pragma unroll
        for (int nb = 0; nb < NT; ++nb) {
            const __bf16* wp = &sW[(nb * 16 + hl) * K + kb0 + off];
            v8bf b0 = *(const v8bf*)(wp);        // ds_load_b128
            v8bf b1 = *(const v8bf*)(wp + 16);   // ds_load_b128
            v16bf b;
#pragma unroll
            for (int j = 0; j < 8; ++j) { b[j] = b0[j]; b[8 + j] = b1[j]; }
            acc[nb] = __builtin_amdgcn_wmma_f32_16x16x32_bf16(
                false, a, false, b, (short)0, acc[nb], false, false);
        }
    }

    const int mlo = (lane < 16) ? 0 : 8;
#pragma unroll
    for (int nb = 0; nb < NT; ++nb) {
        const int n = nb * 16 + hl;
        const float bv = bias ? bias[n] : 0.0f;
#pragma unroll
        for (int v = 0; v < 8; ++v) {
            const int m = row0 + mlo + v;
            float val = acc[nb][v] + bv;
            if (relu) val = fmaxf(val, 0.0f);
            C[(size_t)m * Nout + n] = val;
        }
    }
}

// ---------------------------------------------------------------------------
// Per-node attention coefficients: a_s[n,h] = sum_c h[n,h*32+c]*att_s[h,c]
// ---------------------------------------------------------------------------
__global__ void attn_kernel(const float* __restrict__ h,
                            const float* __restrict__ att_s,
                            const float* __restrict__ att_d,
                            float* __restrict__ a_s, float* __restrict__ a_d,
                            int N)
{
    int idx = blockIdx.x * blockDim.x + threadIdx.x;
    if (idx >= N * 4) return;
    int n = idx >> 2, head = idx & 3;
    const float4* hp = (const float4*)(h + (size_t)n * 128 + head * 32);
    const float4* sp = (const float4*)(att_s + head * 32);
    const float4* dp = (const float4*)(att_d + head * 32);
    float ss = 0.f, sd = 0.f;
#pragma unroll
    for (int j = 0; j < 8; ++j) {
        float4 hv = hp[j], sv = sp[j], dv = dp[j];
        ss += hv.x * sv.x + hv.y * sv.y + hv.z * sv.z + hv.w * sv.w;
        sd += hv.x * dv.x + hv.y * dv.y + hv.z * dv.z + hv.w * dv.w;
    }
    a_s[idx] = ss;
    a_d[idx] = sd;
}

// ---------------------------------------------------------------------------
// Segment-softmax helpers
// ---------------------------------------------------------------------------
__device__ __forceinline__ void atomicMaxF(float* addr, float v)
{
    if (v >= 0.0f) atomicMax((int*)addr, __float_as_int(v));
    else           atomicMin((unsigned int*)addr, __float_as_uint(v));
}

__global__ void init_kernel(float* __restrict__ out, float* __restrict__ mseg,
                            float* __restrict__ sseg, int N)
{
    int i = blockIdx.x * blockDim.x + threadIdx.x;
    if (i < N * 128) out[i] = 0.0f;
    if (i < N * 4) { mseg[i] = -3.402823466e38f; sseg[i] = 0.0f; }
}

__global__ void edge_logits_kernel(const int* __restrict__ ei, int E, int N,
                                   const float* __restrict__ a_s,
                                   const float* __restrict__ a_d,
                                   float* __restrict__ eb,
                                   float* __restrict__ mseg)
{
    int e = blockIdx.x * blockDim.x + threadIdx.x;
    int TOT = E + N;
    if (e >= TOT) return;
    int src = (e < E) ? ei[e]     : (e - E);
    int dst = (e < E) ? ei[E + e] : (e - E);
    float4 as4 = *(const float4*)(a_s + (size_t)src * 4);
    float4 ad4 = *(const float4*)(a_d + (size_t)dst * 4);
    float4 l;
    l.x = as4.x + ad4.x; l.y = as4.y + ad4.y;
    l.z = as4.z + ad4.z; l.w = as4.w + ad4.w;
    l.x = (l.x > 0.f) ? l.x : 0.2f * l.x;
    l.y = (l.y > 0.f) ? l.y : 0.2f * l.y;
    l.z = (l.z > 0.f) ? l.z : 0.2f * l.z;
    l.w = (l.w > 0.f) ? l.w : 0.2f * l.w;
    *(float4*)(eb + (size_t)e * 4) = l;
    float* mp = mseg + (size_t)dst * 4;
    atomicMaxF(mp + 0, l.x);
    atomicMaxF(mp + 1, l.y);
    atomicMaxF(mp + 2, l.z);
    atomicMaxF(mp + 3, l.w);
}

__global__ void edge_exp_kernel(const int* __restrict__ ei, int E, int N,
                                float* __restrict__ eb,
                                const float* __restrict__ mseg,
                                float* __restrict__ sseg)
{
    int e = blockIdx.x * blockDim.x + threadIdx.x;
    int TOT = E + N;
    if (e >= TOT) return;
    int dst = (e < E) ? ei[E + e] : (e - E);
    float4 l  = *(const float4*)(eb + (size_t)e * 4);
    float4 mm = *(const float4*)(mseg + (size_t)dst * 4);
    float4 ex;
    ex.x = __expf(l.x - mm.x);
    ex.y = __expf(l.y - mm.y);
    ex.z = __expf(l.z - mm.z);
    ex.w = __expf(l.w - mm.w);
    *(float4*)(eb + (size_t)e * 4) = ex;
    float* sp = sseg + (size_t)dst * 4;
    atomicAdd(sp + 0, ex.x);
    atomicAdd(sp + 1, ex.y);
    atomicAdd(sp + 2, ex.z);
    atomicAdd(sp + 3, ex.w);
}

// One block (128 threads = 4 waves) per edge; thread f handles feature f.
__global__ __launch_bounds__(128) void scatter_kernel(
    const int* __restrict__ ei, int E, int N,
    const float* __restrict__ h, const float* __restrict__ eb,
    const float* __restrict__ sseg, float* __restrict__ out)
{
    int e = blockIdx.x;
    int f = threadIdx.x;
    int src = (e < E) ? ei[e]     : (e - E);
    int dst = (e < E) ? ei[E + e] : (e - E);
    int head = f >> 5;
    float alpha = eb[(size_t)e * 4 + head] /
                  (sseg[(size_t)dst * 4 + head] + 1e-16f);
    atomicAdd(out + (size_t)dst * 128 + f, h[(size_t)src * 128 + f] * alpha);
}

__global__ void bias_relu_kernel(float* __restrict__ out,
                                 const float* __restrict__ b, int total)
{
    int i = blockIdx.x * blockDim.x + threadIdx.x;
    if (i < total) out[i] = fmaxf(out[i] + b[i & 127], 0.0f);
}

// ---------------------------------------------------------------------------
// Host-side orchestration (all launches on `stream`; ws carved statically)
// ---------------------------------------------------------------------------
extern "C" void kernel_launch(void* const* d_in, const int* in_sizes, int n_in,
                              void* d_out, int out_size, void* d_ws, size_t ws_size,
                              hipStream_t stream)
{
    const float* x  = (const float*)d_in[0];
    const int*   ei = (const int*)d_in[1];
    const float* W[3]  = {(const float*)d_in[2], (const float*)d_in[6],  (const float*)d_in[10]};
    const float* AS[3] = {(const float*)d_in[3], (const float*)d_in[7],  (const float*)d_in[11]};
    const float* AD[3] = {(const float*)d_in[4], (const float*)d_in[8],  (const float*)d_in[12]};
    const float* BB[3] = {(const float*)d_in[5], (const float*)d_in[9],  (const float*)d_in[13]};
    const float* fw1 = (const float*)d_in[14];
    const float* fb1 = (const float*)d_in[15];
    const float* fw2 = (const float*)d_in[16];
    const float* fb2 = (const float*)d_in[17];

    const int N   = in_sizes[0] / 128;
    const int E   = in_sizes[1] / 2;
    const int TOT = E + N;

    // Workspace layout (floats)
    float* buf0 = (float*)d_ws;                    // h  : N*128
    float* buf1 = buf0 + (size_t)N * 128;          // out: N*128
    float* a_s  = buf1 + (size_t)N * 128;          // N*4
    float* a_d  = a_s  + (size_t)N * 4;            // N*4
    float* mseg = a_d  + (size_t)N * 4;            // N*4
    float* sseg = mseg + (size_t)N * 4;            // N*4
    float* eb   = sseg + (size_t)N * 4;            // TOT*4
    float* y1   = eb;                              // reuse (TOT*4 >= N*32)

    const int gemmGrid = (N + 127) / 128;
    const int initGrid = (N * 128 + 255) / 256;
    const int attnGrid = (N * 4 + 255) / 256;
    const int edgeGrid = (TOT + 255) / 256;

    const float* in = x;
    for (int l = 0; l < 3; ++l) {
        // h = in @ W  (GEMM runs first so init may safely clobber buf1==in)
        wmma_gemm<4, 8><<<gemmGrid, 256, 0, stream>>>(in, W[l], nullptr, buf0, N, 0);
        init_kernel<<<initGrid, 256, 0, stream>>>(buf1, mseg, sseg, N);
        attn_kernel<<<attnGrid, 256, 0, stream>>>(buf0, AS[l], AD[l], a_s, a_d, N);
        edge_logits_kernel<<<edgeGrid, 256, 0, stream>>>(ei, E, N, a_s, a_d, eb, mseg);
        edge_exp_kernel<<<edgeGrid, 256, 0, stream>>>(ei, E, N, eb, mseg, sseg);
        scatter_kernel<<<TOT, 128, 0, stream>>>(ei, E, N, buf0, eb, sseg, buf1);
        bias_relu_kernel<<<initGrid, 256, 0, stream>>>(buf1, BB[l], N * 128);
        in = buf1;
    }

    // FC head: y1 = relu(h @ fw1 + fb1); out = y1 @ fw2 + fb2
    wmma_gemm<4, 2><<<gemmGrid, 256, 0, stream>>>(buf1, fw1, fb1, y1, N, 1);
    wmma_gemm<1, 8><<<gemmGrid, 256, 0, stream>>>(y1, fw2, fb2, (float*)d_out, N, 0);
}
